// InformationPlane_79671643341673
// MI455X (gfx1250) — matrix operand order repaired and checked by reading, not compiled, not run on hardware
//
#include <hip/hip_runtime.h>

typedef __attribute__((ext_vector_type(16))) __bf16 v16bf;
typedef __attribute__((ext_vector_type(8)))  float  v8f;

#define NB      8          // number of mini-batches
#define MBS     512        // mini-batch size
#define MM      262144     // 512*512 elements per matrix
#define NSIG    50
#define EPSW    1e-6f

// ---------------- helpers ----------------

__device__ __forceinline__ void split_bf16(float f, __bf16& hi, __bf16& lo) {
    // hi = truncate-to-bf16 (exact as bf16), lo = residual
    float fh = __uint_as_float(__float_as_uint(f) & 0xffff0000u);
    hi = (__bf16)fh;
    lo = (__bf16)(f - fh);
}

__device__ __forceinline__ float block_reduce256(float v, float* sm) {
    int t = threadIdx.x;
    __syncthreads();
    sm[t] = v;
    __syncthreads();
    for (int off = 128; off > 0; off >>= 1) {
        if (t < off) sm[t] += sm[t + off];
        __syncthreads();
    }
    return sm[0];
}

// ---------------- out copy (x passthrough) ----------------

__global__ void copy_x_kernel(const float4* __restrict__ src, float4* __restrict__ dst, int n4) {
    int i = blockIdx.x * blockDim.x + threadIdx.x;
    if (i < n4) dst[i] = src[i];
}

__global__ void init_scal_kernel(float* kyn2) {
    if (threadIdx.x < NB) kyn2[threadIdx.x] = 0.f;
}

// ---------------- WMMA Gram: G[b] = Xb @ Xb^T  (fp32 via bf16 hi/lo split) ----------------

__device__ __forceinline__ void load_frag_A(const float* __restrict__ row, int k0, int half,
                                            v16bf& hi, v16bf& lo) {
    // A 16x32 bf16 frag, contiguous k order used here: lane<16 -> K = k0+0..15 ; lane>=16 -> K = k0+16..31
    const float* p0 = row + k0 + half * 16;
    const float* p1 = row + k0 + half * 16 + 8;
#pragma unroll
    for (int e = 0; e < 8; ++e) { __bf16 h, l; split_bf16(p0[e], h, l); hi[e] = h; lo[e] = l; }
#pragma unroll
    for (int e = 0; e < 8; ++e) { __bf16 h, l; split_bf16(p1[e], h, l); hi[8 + e] = h; lo[8 + e] = l; }
}

__device__ __forceinline__ void load_frag_B(const float* __restrict__ row, int k0, int half,
                                            v16bf& hi, v16bf& lo) {
    // B 32x16 bf16 frag, contiguous k order used here: lane<16 -> K = k0+0..15 ; lane>=16 -> K = k0+16..31
    const float* p = row + k0 + half * 16;
#pragma unroll
    for (int e = 0; e < 16; ++e) { __bf16 h, l; split_bf16(p[e], h, l); hi[e] = h; lo[e] = l; }
}

__global__ void gram_wmma_kernel(const float* __restrict__ X, int K, float* __restrict__ G) {
    const int lane = threadIdx.x;       // one wave per block
    const int half = lane >> 4;
    const int r    = lane & 15;
    const int tn = blockIdx.x, tm = blockIdx.y, b = blockIdx.z;

    const float* rowA = X + (size_t)(b * MBS + tm * 16 + r) * (size_t)K;
    const float* rowB = X + (size_t)(b * MBS + tn * 16 + r) * (size_t)K;

    v8f acc = {0.f, 0.f, 0.f, 0.f, 0.f, 0.f, 0.f, 0.f};

    for (int k0 = 0; k0 < K; k0 += 32) {
        if (k0 + 64 < K) {   // keep data streaming through WGP$/L2
            __builtin_prefetch(rowA + k0 + 64, 0, 3);
            __builtin_prefetch(rowB + k0 + 64, 0, 3);
        }
        v16bf ah, al, bh, bl;
        load_frag_A(rowA, k0, half, ah, al);
        load_frag_B(rowB, k0, half, bh, bl);
        // fp32-accurate product: hi*hi + hi*lo + lo*hi (lo*lo negligible)
        acc = __builtin_amdgcn_wmma_f32_16x16x32_bf16(false, ah, false, bh, (short)0, acc, false, false);
        acc = __builtin_amdgcn_wmma_f32_16x16x32_bf16(false, ah, false, bl, (short)0, acc, false, false);
        acc = __builtin_amdgcn_wmma_f32_16x16x32_bf16(false, al, false, bh, (short)0, acc, false, false);
    }

    float* Gb = G + (size_t)b * MM;
    const int col = tn * 16 + r;
#pragma unroll
    for (int v = 0; v < 8; ++v)
        Gb[(size_t)(tm * 16 + v + half * 8) * MBS + col] = acc[v];
}

// ---------------- diagonals (read before Gram buffers get overwritten) ----------------

__global__ void diag_kernel(const float* __restrict__ GX, const float* __restrict__ GI,
                            float* __restrict__ dgx, float* __restrict__ dgi) {
    int b = blockIdx.x, i = threadIdx.x;  // 512 threads
    size_t o = (size_t)b * MM + (size_t)i * MBS + i;
    dgx[b * MBS + i] = GX[o];
    dgi[b * MBS + i] = GI[o];
}

// ---------------- label kernel Ky = exp(-d2/0.01), plus ||Ky||_F^2 ----------------

__global__ void ky_kernel(const float* __restrict__ lab, float* __restrict__ KY,
                          float* __restrict__ kyn2) {
    __shared__ float sm[256];
    int b = blockIdx.y;
    int idx = blockIdx.x * 256 + threadIdx.x;           // 0..262143
    int i = idx >> 9, j = idx & 511;
    const float* li = lab + (size_t)(b * MBS + i) * 10;
    const float* lj = lab + (size_t)(b * MBS + j) * 10;
    float di = 0.f, dj = 0.f, dot = 0.f;
#pragma unroll
    for (int u = 0; u < 10; ++u) {
        float a = li[u], c = lj[u];
        di += a * a; dj += c * c; dot += a * c;
    }
    float d2 = fmaxf(di + dj - 2.f * dot, 0.f);
    float ky = __expf(-d2 * 100.f);                     // sigma = 0.1
    KY[(size_t)b * MM + idx] = ky;
    float p = block_reduce256(ky * ky, sm);
    if (threadIdx.x == 0) atomicAdd(&kyn2[b], p);
}

// ---------------- mean nonzero distance per batch ----------------

__global__ void stats_kernel(const float* __restrict__ GX, const float* __restrict__ dgx,
                             float* __restrict__ meanD) {
    __shared__ float sm[256];
    int b = blockIdx.x;
    const float* Gb = GX + (size_t)b * MM;
    const float* D  = dgx + b * MBS;
    float sd = 0.f, cnt = 0.f;
    for (int idx = threadIdx.x; idx < MM; idx += 256) {
        int i = idx >> 9, j = idx & 511;
        float d2 = fmaxf(D[i] + D[j] - 2.f * Gb[idx], 0.f);
        sd  += sqrtf(d2);
        cnt += (d2 > 0.f) ? 1.f : 0.f;
    }
    float tsd = block_reduce256(sd, sm);
    float tc  = block_reduce256(cnt, sm);
    if (threadIdx.x == 0) meanD[b] = tsd / fmaxf(tc, 1.f);
}

// ---------------- kernel-alignment sums for 50 sigmas ----------------

__global__ void align_kernel(const float* __restrict__ GX, const float* __restrict__ KY,
                             const float* __restrict__ dgx, const float* __restrict__ meanD,
                             float* __restrict__ num, float* __restrict__ den2) {
    __shared__ float sm[256];
    int s = blockIdx.x, b = blockIdx.y;
    float m = meanD[b];
    float start = 0.1f * m;
    float step  = (10.f * m - start) * (1.f / (float)NSIG);
    float sig   = start + step * (float)s;
    float inv   = 1.f / (sig * sig);
    const float* Gb = GX + (size_t)b * MM;
    const float* Kb = KY + (size_t)b * MM;
    const float* D  = dgx + b * MBS;
    float sn = 0.f, sd = 0.f;
    for (int idx = threadIdx.x; idx < MM; idx += 256) {
        int i = idx >> 9, j = idx & 511;
        float d2 = fmaxf(D[i] + D[j] - 2.f * Gb[idx], 0.f);
        float ks = __expf(-d2 * inv);
        sn += ks * Kb[idx];
        sd += ks * ks;
    }
    float tn_ = block_reduce256(sn, sm);
    float td  = block_reduce256(sd, sm);
    if (threadIdx.x == 0) { num[b * NSIG + s] = tn_; den2[b * NSIG + s] = td; }
}

// ---------------- sigma argmax + sequential EMA (beta = 0.5) ----------------

__global__ void select_kernel(const float* __restrict__ num, const float* __restrict__ den2,
                              const float* __restrict__ kyn2, const float* __restrict__ meanD,
                              float* __restrict__ sigE) {
    if (threadIdx.x != 0 || blockIdx.x != 0) return;
    float prev = 0.f;
    for (int b = 0; b < NB; ++b) {
        float m = meanD[b];
        float start = 0.1f * m;
        float step  = (10.f * m - start) * (1.f / (float)NSIG);
        float kn = sqrtf(kyn2[b]);
        float best = -3.4e38f, bsig = start;
        for (int s = 0; s < NSIG; ++s) {
            float loss = num[b * NSIG + s] / (sqrtf(den2[b * NSIG + s]) * kn);
            if (loss > best) { best = loss; bsig = start + step * (float)s; }
        }
        prev = (b == 0) ? bsig : 0.5f * prev + 0.5f * bsig;   // EMA carried across batches
        sigE[b] = prev;
    }
}

// ---------------- build the 5 matrices per batch (in place where possible) ----------------
// trace identities: diag(A)=diag(Ax)=diag(Ay)=1/n exactly, so trace(Ax*A)=trace(A*Ay)=1/n
// => n1 = Ax.A*n ; n2 = A.Ay*n = A.Ky

__global__ void build_kernel(float* __restrict__ GX, float* __restrict__ GI, float* __restrict__ KY,
                             float* __restrict__ N1, float* __restrict__ N2,
                             const float* __restrict__ dgx, const float* __restrict__ dgi,
                             const float* __restrict__ sigE) {
    int b = blockIdx.y;
    int idx = blockIdx.x * 256 + threadIdx.x;
    int i = idx >> 9, j = idx & 511;
    size_t o = (size_t)b * MM + idx;
    float sg = sigE[b];
    float invx = 1.f / (sg * sg);
    float d2x = fmaxf(dgx[b * MBS + i] + dgx[b * MBS + j] - 2.f * GX[o], 0.f);
    float d2i = fmaxf(dgi[b * MBS + i] + dgi[b * MBS + j] - 2.f * GI[o], 0.f);
    float a   = __expf(-d2x * invx)        * (1.f / (float)MBS);
    float ax  = __expf(-d2i * (1.f / 64.f)) * (1.f / (float)MBS);   // sigma=8
    float ky  = KY[o];
    GX[o] = a;                        // A
    GI[o] = ax;                       // Ax
    KY[o] = ky * (1.f / (float)MBS);  // Ay
    N1[o] = ax * a * (float)MBS;      // (Ax*A)/trace
    N2[o] = a * ky;                   // (A*Ay)/trace
}

// ---------------- batched one-sided (Hestenes) Jacobi eigensolver + entropy ----------------
// One workgroup per matrix (40). PSD => singular values == eigenvalues.
// Round-robin tournament: 511 rounds of 256 disjoint pairs cover all C(512,2) pairs.

__global__ void jacobi_kernel(float* __restrict__ GX, float* __restrict__ GI, float* __restrict__ KY,
                              float* __restrict__ N1, float* __restrict__ N2,
                              float* __restrict__ ent) {
    __shared__ float sm[256];
    int m = blockIdx.x;                 // 0..39
    int b = m / 5, w = m % 5;
    float* base = (w == 0) ? GX : (w == 1) ? GI : (w == 2) ? KY : (w == 3) ? N1 : N2;
    float* M = base + (size_t)b * MM;
    int tid = threadIdx.x;

    for (int sweep = 0; sweep < 8; ++sweep) {
        for (int r = 0; r < 511; ++r) {
            int k0 = tid;                    // 0..255
            int k1 = 511 - tid;              // 256..511 (never position 0)
            int p = (k0 == 0) ? 0 : 1 + ((k0 - 1 + r) % 511);
            int q = 1 + ((k1 - 1 + r) % 511);
            float* va = M + (size_t)p * MBS;
            float* vb = M + (size_t)q * MBS;
            float alpha = 0.f, beta = 0.f, gamma = 0.f;
#pragma unroll 4
            for (int k = 0; k < MBS; ++k) {
                float x = va[k], y = vb[k];
                alpha += x * x; beta += y * y; gamma += x * y;
            }
            if (fabsf(gamma) > 1e-30f) {
                float zeta = (beta - alpha) / (2.f * gamma);
                float t = copysignf(1.f / (fabsf(zeta) + sqrtf(1.f + zeta * zeta)), zeta);
                float c = rsqrtf(1.f + t * t);
                float s = c * t;
#pragma unroll 4
                for (int k = 0; k < MBS; ++k) {
                    float x = va[k], y = vb[k];
                    va[k] = c * x - s * y;
                    vb[k] = s * x + c * y;
                }
            }
            __syncthreads();
        }
    }

    // eigenvalues = row norms; entropy = -sum (w+eps) log2 (w+eps)
    float h = 0.f;
    for (int j = tid; j < MBS; j += 256) {
        const float* row = M + (size_t)j * MBS;
        float s2 = 0.f;
#pragma unroll 4
        for (int k = 0; k < MBS; ++k) s2 += row[k] * row[k];
        float wv = sqrtf(s2) + EPSW;
        h -= wv * log2f(wv);
    }
    float th = block_reduce256(h, sm);
    if (tid == 0) ent[m] = th;
}

// ---------------- combine entropies into the two MI scalars ----------------

__global__ void finalize_kernel(const float* __restrict__ ent, float* __restrict__ out) {
    if (threadIdx.x != 0 || blockIdx.x != 0) return;
    float ixt = 0.f, ity = 0.f;
    for (int b = 0; b < NB; ++b) {
        const float* e = ent + b * 5;       // 0:A 1:Ax 2:Ay 3:n1 4:n2
        ixt += e[1] + e[0] - e[3];          // H(Ax)+H(A)-H(joint)
        ity += e[0] + e[2] - e[4];          // H(A)+H(Ay)-H(joint)
    }
    const size_t NX = 33554432ull;          // 4096*8192
    out[NX]     = ixt * (1.f / (float)NB);
    out[NX + 1] = ity * (1.f / (float)NB);
}

// ---------------- launch ----------------

extern "C" void kernel_launch(void* const* d_in, const int* in_sizes, int n_in,
                              void* d_out, int out_size, void* d_ws, size_t ws_size,
                              hipStream_t stream) {
    (void)in_sizes; (void)n_in; (void)out_size; (void)ws_size;
    const float* x   = (const float*)d_in[0];   // (4096, 8192)
    const float* inp = (const float*)d_in[1];   // (4096, 3*32*32=3072)
    const float* lab = (const float*)d_in[2];   // (4096, 10)
    float* out = (float*)d_out;
    float* ws  = (float*)d_ws;

    const size_t MATS = (size_t)NB * MM;        // 2,097,152 floats per region
    float* GX  = ws;                            // Gram(x)  -> A
    float* GI  = GX + MATS;                     // Gram(in) -> Ax
    float* KY  = GI + MATS;                     // Ky       -> Ay
    float* N1  = KY + MATS;                     // (Ax*A)/tr
    float* N2  = N1 + MATS;                     // (A*Ay)/tr
    float* scal  = N2 + MATS;
    float* dgx   = scal;                        // 4096
    float* dgi   = dgx + 4096;                  // 4096
    float* kyn2  = dgi + 4096;                  // 8
    float* meanD = kyn2 + 8;                    // 8
    float* num   = meanD + 8;                   // 400
    float* den2  = num + 400;                   // 400
    float* sigE  = den2 + 400;                  // 8
    float* ent   = sigE + 8;                    // 40

    // 1. pass x through to out (float4, HBM-bound)
    copy_x_kernel<<<(8388608 + 255) / 256, 256, 0, stream>>>((const float4*)x, (float4*)out, 8388608);
    // 2. zero atomic accumulators (ws is not re-poisoned between replays)
    init_scal_kernel<<<1, 64, 0, stream>>>(kyn2);
    // 3. WMMA Gram matrices (bf16 hi/lo split, 3 wmma per 32-K step)
    gram_wmma_kernel<<<dim3(32, 32, NB), 32, 0, stream>>>(x,   8192, GX);
    gram_wmma_kernel<<<dim3(32, 32, NB), 32, 0, stream>>>(inp, 3072, GI);
    // 4. capture diagonals before in-place transforms
    diag_kernel<<<NB, 512, 0, stream>>>(GX, GI, dgx, dgi);
    // 5. label kernel + Frobenius norm
    ky_kernel<<<dim3(1024, NB), 256, 0, stream>>>(lab, KY, kyn2);
    // 6. mean nonzero pairwise distance per batch
    stats_kernel<<<NB, 256, 0, stream>>>(GX, dgx, meanD);
    // 7. kernel-alignment sums over 50 sigmas
    align_kernel<<<dim3(NSIG, NB), 256, 0, stream>>>(GX, KY, dgx, meanD, num, den2);
    // 8. argmax sigma + cross-batch EMA scan (scalar, sequential by definition)
    select_kernel<<<1, 32, 0, stream>>>(num, den2, kyn2, meanD, sigE);
    // 9. build A, Ax, Ay, n1, n2
    build_kernel<<<dim3(1024, NB), 256, 0, stream>>>(GX, GI, KY, N1, N2, dgx, dgi, sigE);
    // 10. 40 batched Jacobi eigensolves + Renyi entropies
    jacobi_kernel<<<NB * 5, 256, 0, stream>>>(GX, GI, KY, N1, N2, ent);
    // 11. MI means -> out[N], out[N+1]
    finalize_kernel<<<1, 32, 0, stream>>>(ent, out);
}
